// GNNEncoder_23940147708101
// MI455X (gfx1250) — matrix operand order repaired and checked by reading, not compiled, not run on hardware
//
#include <hip/hip_runtime.h>

typedef __attribute__((ext_vector_type(16))) _Float16 v16h;
typedef __attribute__((ext_vector_type(8)))  float    v8f;

constexpr int NN = 16384;      // nodes
constexpr int EE = 262144;     // edges
constexpr int HD = 128;        // hidden
constexpr int NH = 4;          // heads
constexpr int EN = EE + NN;    // edges + self loops
constexpr int BB = 16;         // graphs
constexpr int NPG = NN / BB;   // 1024 nodes per graph

// ---------------- helpers ----------------
__device__ __forceinline__ float warp_sum(float x) {
#pragma unroll
  for (int off = 16; off > 0; off >>= 1) x += __shfl_xor(x, off, 32);
  return x;
}
__device__ __forceinline__ float sigf(float x) { return 1.f / (1.f + expf(-x)); }

// monotonic float<->uint map for atomicMax on floats (incl. negatives)
__device__ __forceinline__ unsigned fmap(float f) {
  unsigned u = __float_as_uint(f);
  return (u & 0x80000000u) ? ~u : (u | 0x80000000u);
}
__device__ __forceinline__ float funmap(unsigned u) {
  return __uint_as_float((u & 0x80000000u) ? (u ^ 0x80000000u) : ~u);
}

// recompute one l2-normalized edge-embedding row (lane holds 4 of 128 cols)
__device__ __forceinline__ void edge_row(const float* __restrict__ ea4,
                                         const float* __restrict__ W,
                                         const float* __restrict__ b,
                                         int lane, float v[4]) {
  float e0 = ea4[0], e1 = ea4[1], e2 = ea4[2], e3 = ea4[3];
  float ss = 0.f;
#pragma unroll
  for (int c = 0; c < 4; ++c) {
    int col = lane * 4 + c;
    float t = e0 * W[col] + e1 * W[128 + col] + e2 * W[256 + col] + e3 * W[384 + col] + b[col];
    v[c] = t; ss += t * t;
  }
  ss = warp_sum(ss);
  float sc = 1.f / fmaxf(sqrtf(ss), 1e-12f);
#pragma unroll
  for (int c = 0; c < 4; ++c) v[c] *= sc;
}

// A-fragment swizzle: (m, K) -> (lane, i), contiguous k order: hi = K>>4, i = K&15 (matches B)
__device__ __forceinline__ void a_swz(int K, int& hi, int& i) {
  hi = K >> 4; i = K & 15;
}

// ---------------- stage 0: embeddings ----------------
__global__ __launch_bounds__(256) void node_embed_kernel(
    const float* __restrict__ x, const float* __restrict__ w,
    const float* __restrict__ b, float* __restrict__ h) {
  int idx = blockIdx.x * 256 + threadIdx.x;        // NN*128
  int n = idx >> 7, d = idx & 127;
  float acc = b[d];
  const float* xr = x + n * 8;
#pragma unroll
  for (int k = 0; k < 8; ++k) acc += xr[k] * w[k * 128 + d];
  h[idx] = acc;
}

__global__ __launch_bounds__(256) void deg_loop_kernel(
    const float* __restrict__ edge_attr, const float* __restrict__ ew,
    const float* __restrict__ eb, const int* __restrict__ ei,
    float* __restrict__ loop_attr, float* __restrict__ deg) {
  int wid = blockIdx.x * 8 + (threadIdx.x >> 5);   // one wave per edge
  int lane = threadIdx.x & 31;
  float v[4];
  edge_row(edge_attr + (size_t)wid * 4, ew, eb, lane, v);
  int d = ei[EE + wid];
#pragma unroll
  for (int c = 0; c < 4; ++c) atomicAdd(&loop_attr[(size_t)d * 128 + lane * 4 + c], v[c]);
  if (lane == 0) atomicAdd(&deg[d], 1.f);
}

__global__ __launch_bounds__(256) void loop_div_kernel(float* __restrict__ loop_attr,
                                                       const float* __restrict__ deg) {
  int idx = blockIdx.x * 256 + threadIdx.x;        // NN*128
  loop_attr[idx] /= fmaxf(deg[idx >> 7], 1.f);
}

// ---------------- per-layer prep ----------------
// convert lin_w[l] (f32 [128,512]) to f16 pre-swizzled into WMMA B-fragment order:
// dst[((ntile*4 + kc)*32 + lane)*16 + i], lane = n%16 + 16*(K>>4), i = K&15, K = k&31
__global__ __launch_bounds__(256) void conv_b_swz_kernel(const float* __restrict__ src,
                                                         _Float16* __restrict__ dst) {
  int idx = blockIdx.x * 256 + threadIdx.x;        // 128*512
  int k = idx >> 9, n = idx & 511;
  int ntile = n >> 4, nr = n & 15;
  int kc = k >> 5, K = k & 31;
  int hi = K >> 4, i = K & 15;
  int lane = nr + 16 * hi;
  dst[(size_t)((ntile * 4 + kc) * 32 + lane) * 16 + i] = (_Float16)src[idx];
}

// watt[k][head] = sum_d lin_edge_w[k, head*128+d] * att_edge[head, d]
__global__ __launch_bounds__(256) void fold_watt_kernel(const float* __restrict__ lew,
                                                        const float* __restrict__ ae,
                                                        float* __restrict__ watt) {
  int idx = blockIdx.x * 256 + threadIdx.x;        // 512
  int k = idx >> 2, head = idx & 3;
  float acc = 0.f;
  const float* lw = lew + (size_t)k * 512 + head * 128;
  const float* av = ae + head * 128;
  for (int d = 0; d < 128; ++d) acc += lw[d] * av[d];
  watt[k * 4 + head] = acc;
}

// ---------------- WMMA GEMM: xh = h[N,128] @ lin_w[128,512] ----------------
// Bh is pre-swizzled (fragment order). A tile is swizzled into LDS fragment order.
__global__ __launch_bounds__(256) void gemm_xh_kernel(const float* __restrict__ A,
                                                      const _Float16* __restrict__ Bh,
                                                      float* __restrict__ C) {
  __shared__ _Float16 As[16 * 128];   // fragment order: [(kc*32 + lane)*16 + i]
  int m0 = blockIdx.x * 16;
  for (int idx = threadIdx.x; idx < 16 * 128; idx += 256) {
    int m = idx >> 7, c = idx & 127;          // c = global K
    int kc = c >> 5, K = c & 31;
    int hi, i;
    a_swz(K, hi, i);
    int lane = m + 16 * hi;
    As[(kc * 32 + lane) * 16 + i] = (_Float16)A[(size_t)(m0 + m) * 128 + c];
  }
  __syncthreads();

  int wave = threadIdx.x >> 5;   // 0..7, each wave does 4 N-tiles
  int lane = threadIdx.x & 31;
  int nr = lane & 15;
  int hi = lane >> 4;

  // A fragments are shared by all 4 N-tiles: load once
  v16h a[4];
#pragma unroll
  for (int kc = 0; kc < 4; ++kc)
    a[kc] = *(const v16h*)&As[(kc * 32 + lane) * 16];

#pragma unroll
  for (int t = 0; t < 4; ++t) {
    int ntile = wave * 4 + t;
    const _Float16* bp = Bh + (size_t)(ntile * 4) * 32 * 16 + (size_t)lane * 16;
    v8f acc = {};
#pragma unroll
    for (int kc = 0; kc < 4; ++kc) {
      v16h b = *(const v16h*)(bp + (size_t)kc * 32 * 16);
      acc = __builtin_amdgcn_wmma_f32_16x16x32_f16(false, a[kc], false, b, (short)0, acc,
                                                   false, false);
    }
    int col0 = ntile * 16;
#pragma unroll
    for (int r = 0; r < 8; ++r)
      C[(size_t)(m0 + r + 8 * hi) * 512 + col0 + nr] = acc[r];
  }
}

// ---------------- attention scores ----------------
__global__ __launch_bounds__(256) void attn_node_kernel(const float* __restrict__ xh,
                                                        const float* __restrict__ asv,
                                                        const float* __restrict__ adv,
                                                        float* __restrict__ a_src,
                                                        float* __restrict__ a_dst) {
  int n = blockIdx.x * 8 + (threadIdx.x >> 5);     // one wave per node
  int lane = threadIdx.x & 31;
  const float* xr = xh + (size_t)n * 512;
#pragma unroll
  for (int head = 0; head < 4; ++head) {
    float ps = 0.f, pd = 0.f;
#pragma unroll
    for (int j = 0; j < 4; ++j) {
      int d = lane + j * 32;
      float v = xr[head * 128 + d];
      ps += v * asv[head * 128 + d];
      pd += v * adv[head * 128 + d];
    }
    ps = warp_sum(ps); pd = warp_sum(pd);
    if (lane == 0) { a_src[n * 4 + head] = ps; a_dst[n * 4 + head] = pd; }
  }
}

__global__ __launch_bounds__(256) void a_edge_kernel(const float* __restrict__ edge_attr,
                                                     const float* __restrict__ ew,
                                                     const float* __restrict__ eb,
                                                     const float* __restrict__ loop_attr,
                                                     const float* __restrict__ watt,
                                                     float* __restrict__ a_edge) {
  int wid = blockIdx.x * 8 + (threadIdx.x >> 5);   // one wave per (edge or loop)
  int lane = threadIdx.x & 31;
  float v[4];
  if (wid < EE) {
    edge_row(edge_attr + (size_t)wid * 4, ew, eb, lane, v);
  } else {
    const float* lr = loop_attr + (size_t)(wid - EE) * 128;
#pragma unroll
    for (int c = 0; c < 4; ++c) v[c] = lr[lane * 4 + c];
  }
#pragma unroll
  for (int head = 0; head < 4; ++head) {
    float p = 0.f;
#pragma unroll
    for (int c = 0; c < 4; ++c) p += v[c] * watt[(lane * 4 + c) * 4 + head];
    p = warp_sum(p);
    if (lane == 0) a_edge[wid * 4 + head] = p;
  }
}

// ---------------- segment softmax + aggregation ----------------
__global__ __launch_bounds__(256) void alpha_max_kernel(const float* __restrict__ a_src,
                                                        const float* __restrict__ a_dst,
                                                        const float* __restrict__ a_edge,
                                                        const int* __restrict__ ei,
                                                        float* __restrict__ alpha,
                                                        unsigned* __restrict__ mx) {
  int idx = blockIdx.x * 256 + threadIdx.x;        // EN*4
  int ef = idx >> 2, head = idx & 3;
  int sf, df;
  if (ef < EE) { sf = ei[ef]; df = ei[EE + ef]; } else { sf = df = ef - EE; }
  float al = a_src[sf * 4 + head] + a_dst[df * 4 + head] + a_edge[idx];
  al = al > 0.f ? al : 0.2f * al;                  // leaky relu
  alpha[idx] = al;
  atomicMax(&mx[df * 4 + head], fmap(al));
}

__global__ __launch_bounds__(256) void exp_sum_kernel(float* __restrict__ alpha,
                                                      const unsigned* __restrict__ mx,
                                                      const int* __restrict__ ei,
                                                      float* __restrict__ denom) {
  int idx = blockIdx.x * 256 + threadIdx.x;        // EN*4
  int ef = idx >> 2, head = idx & 3;
  int df = (ef < EE) ? ei[EE + ef] : ef - EE;
  float ex = expf(alpha[idx] - funmap(mx[df * 4 + head]));
  alpha[idx] = ex;
  atomicAdd(&denom[df * 4 + head], ex);
}

__global__ __launch_bounds__(256) void aggregate_kernel(const float* __restrict__ ex,
                                                        const float* __restrict__ denom,
                                                        const float* __restrict__ xh,
                                                        const int* __restrict__ ei,
                                                        float* __restrict__ agg) {
  int wid = blockIdx.x * 8 + (threadIdx.x >> 5);   // one wave per (edge, head)
  int lane = threadIdx.x & 31;
  int ef = wid >> 2, head = wid & 3;
  int sf, df;
  if (ef < EE) { sf = ei[ef]; df = ei[EE + ef]; } else { sf = df = ef - EE; }
  float w = ex[ef * 4 + head] / denom[df * 4 + head];
  const float4 xv = *(const float4*)(xh + (size_t)sf * 512 + head * 128 + lane * 4);
  float* dst = agg + (size_t)df * 512 + head * 128 + lane * 4;
  atomicAdd(dst + 0, w * xv.x);
  atomicAdd(dst + 1, w * xv.y);
  atomicAdd(dst + 2, w * xv.z);
  atomicAdd(dst + 3, w * xv.w);
}

// ---------------- head mean + bias, batchnorm, residual ----------------
__global__ __launch_bounds__(256) void finalize_kernel(const float* __restrict__ agg,
                                                       const float* __restrict__ lb,
                                                       float* __restrict__ outb) {
  int idx = blockIdx.x * 256 + threadIdx.x;        // NN*128
  int n = idx >> 7, d = idx & 127;
  const float* a = agg + (size_t)n * 512 + d;
  outb[idx] = 0.25f * (a[0] + a[128] + a[256] + a[384]) + lb[d];
}

__global__ __launch_bounds__(128) void bn_stats_kernel(const float* __restrict__ outb,
                                                       float* __restrict__ s,
                                                       float* __restrict__ s2) {
  int d = threadIdx.x;
  int base = blockIdx.x * 128;                     // 128 blocks x 128 rows
  float a = 0.f, b = 0.f;
  for (int r = 0; r < 128; ++r) {
    float x = outb[(size_t)(base + r) * 128 + d];
    a += x; b += x * x;
  }
  atomicAdd(&s[d], a);
  atomicAdd(&s2[d], b);
}

__global__ __launch_bounds__(256) void bn_apply_kernel(const float* __restrict__ outb,
                                                       const float* __restrict__ s,
                                                       const float* __restrict__ s2,
                                                       const float* __restrict__ gamma,
                                                       const float* __restrict__ beta,
                                                       float* __restrict__ h) {
  int idx = blockIdx.x * 256 + threadIdx.x;        // NN*128
  int d = idx & 127;
  float mu = s[d] * (1.f / NN);
  float var = s2[d] * (1.f / NN) - mu * mu;
  float y = gamma[d] * (outb[idx] - mu) * rsqrtf(var + 1e-5f) + beta[d];
  h[idx] = fmaxf(y, 0.f) + h[idx];
}

// ---------------- Set2Set ----------------
__global__ __launch_bounds__(512) void lstm_kernel(const float* __restrict__ wih,
                                                   const float* __restrict__ whh,
                                                   const float* __restrict__ lb,
                                                   float* __restrict__ qstar,
                                                   float* __restrict__ hh,
                                                   float* __restrict__ cc) {
  int b = blockIdx.x, t = threadIdx.x;
  __shared__ float qs[256], hs[128], gates[512];
  if (t < 256) qs[t] = qstar[b * 256 + t];
  else if (t < 384) hs[t - 256] = hh[b * 128 + (t - 256)];
  __syncthreads();
  float acc = lb[t];
  const float* wi = wih + (size_t)t * 256;
  for (int k = 0; k < 256; ++k) acc += qs[k] * wi[k];
  const float* wh = whh + (size_t)t * 128;
  for (int k = 0; k < 128; ++k) acc += hs[k] * wh[k];
  gates[t] = acc;
  __syncthreads();
  if (t < 128) {
    float ig = sigf(gates[t]);
    float fg = sigf(gates[128 + t]);
    float gg = tanhf(gates[256 + t]);
    float og = sigf(gates[384 + t]);
    float c = fg * cc[b * 128 + t] + ig * gg;
    cc[b * 128 + t] = c;
    float hv = og * tanhf(c);
    hh[b * 128 + t] = hv;
    qstar[b * 256 + t] = hv;
  }
}

__global__ __launch_bounds__(128) void s2s_attn_kernel(const float* __restrict__ h,
                                                       const float* __restrict__ hh,
                                                       float* __restrict__ qstar) {
  int b = blockIdx.x, t = threadIdx.x;
  __shared__ float hs[128];
  __shared__ float es[NPG];
  __shared__ float red[128];
  hs[t] = hh[b * 128 + t];
  __syncthreads();
  float lmax = -3.4e38f;
  for (int it = 0; it < NPG / 128; ++it) {
    int ln = it * 128 + t;
    const float* hr = h + (size_t)(b * NPG + ln) * 128;
    float e = 0.f;
    for (int k = 0; k < 128; ++k) e += hr[k] * hs[k];
    es[ln] = e;
    lmax = fmaxf(lmax, e);
  }
  red[t] = lmax; __syncthreads();
  for (int s = 64; s; s >>= 1) { if (t < s) red[t] = fmaxf(red[t], red[t + s]); __syncthreads(); }
  float mx = red[0]; __syncthreads();
  float lsum = 0.f;
  for (int it = 0; it < NPG / 128; ++it) {
    int ln = it * 128 + t;
    float e = expf(es[ln] - mx);
    es[ln] = e; lsum += e;
  }
  red[t] = lsum; __syncthreads();
  for (int s = 64; s; s >>= 1) { if (t < s) red[t] += red[t + s]; __syncthreads(); }
  float inv = 1.f / red[0];
  __syncthreads();
  float r = 0.f;
  for (int ln = 0; ln < NPG; ++ln) r += es[ln] * h[(size_t)(b * NPG + ln) * 128 + t];
  qstar[b * 256 + 128 + t] = r * inv;
}

__global__ __launch_bounds__(256) void out_mlp_kernel(const float* __restrict__ qstar,
                                                      const float* __restrict__ w1,
                                                      const float* __restrict__ b1,
                                                      const float* __restrict__ w2,
                                                      const float* __restrict__ b2,
                                                      float* __restrict__ out) {
  int b = blockIdx.x, t = threadIdx.x;
  __shared__ float qs[256], g1[128], red[256];
  qs[t] = qstar[b * 256 + t];
  __syncthreads();
  if (t < 128) {
    float a = b1[t];
    for (int k = 0; k < 256; ++k) a += qs[k] * w1[k * 128 + t];
    g1[t] = fmaxf(a, 0.f);
  }
  __syncthreads();
  float a = b2[t];
  for (int k = 0; k < 128; ++k) a += g1[k] * w2[k * 256 + t];
  float g2 = fmaxf(a, 0.f);
  red[t] = g2 * g2; __syncthreads();
  for (int s = 128; s; s >>= 1) { if (t < s) red[t] += red[t + s]; __syncthreads(); }
  float nrm = sqrtf(red[0]);
  out[b * 256 + t] = g2 / fmaxf(nrm, 1e-12f);
}

// ---------------- host ----------------
extern "C" void kernel_launch(void* const* d_in, const int* in_sizes, int n_in,
                              void* d_out, int out_size, void* d_ws, size_t ws_size,
                              hipStream_t stream) {
  const float* x         = (const float*)d_in[0];
  const float* edge_attr = (const float*)d_in[1];
  const float* node_w    = (const float*)d_in[2];
  const float* node_b    = (const float*)d_in[3];
  const float* edge_w    = (const float*)d_in[4];
  const float* edge_b    = (const float*)d_in[5];
  const float* lin_w     = (const float*)d_in[6];
  const float* lin_b     = (const float*)d_in[7];
  const float* lin_edge_w= (const float*)d_in[8];
  const float* att_src   = (const float*)d_in[9];
  const float* att_dst   = (const float*)d_in[10];
  const float* att_edge  = (const float*)d_in[11];
  const float* bn_gamma  = (const float*)d_in[12];
  const float* bn_beta   = (const float*)d_in[13];
  const float* lstm_wih  = (const float*)d_in[14];
  const float* lstm_whh  = (const float*)d_in[15];
  const float* lstm_b    = (const float*)d_in[16];
  const float* out_w1    = (const float*)d_in[17];
  const float* out_b1    = (const float*)d_in[18];
  const float* out_w2    = (const float*)d_in[19];
  const float* out_b2    = (const float*)d_in[20];
  const int*   edge_index= (const int*)d_in[21];
  float* out = (float*)d_out;

  // workspace carve-up
  char* p = (char*)d_ws;
  size_t off = 0;
  auto take = [&](size_t bytes) { char* q = p + off; off = (off + bytes + 255) & ~(size_t)255; return q; };
  float*    h         = (float*)take((size_t)NN * 128 * 4);
  float*    loop_attr = (float*)take((size_t)NN * 128 * 4);
  float*    deg       = (float*)take((size_t)NN * 4);
  _Float16* Bh        = (_Float16*)take((size_t)128 * 512 * 2);
  float*    watt      = (float*)take(512 * 4);
  float*    xh        = (float*)take((size_t)NN * 512 * 4);
  float*    agg       = (float*)take((size_t)NN * 512 * 4);
  float*    a_src     = (float*)take((size_t)NN * 4 * 4);
  float*    a_dst     = (float*)take((size_t)NN * 4 * 4);
  float*    a_edge    = (float*)take((size_t)EN * 4 * 4);
  float*    alpha     = (float*)take((size_t)EN * 4 * 4);
  unsigned* mx        = (unsigned*)take((size_t)NN * 4 * 4);
  float*    denom     = (float*)take((size_t)NN * 4 * 4);
  float*    outb      = (float*)take((size_t)NN * 128 * 4);
  float*    bnsum     = (float*)take(128 * 4);
  float*    bnsq      = (float*)take(128 * 4);
  float*    qstar     = (float*)take(BB * 256 * 4);
  float*    hhb       = (float*)take(BB * 128 * 4);
  float*    ccb       = (float*)take(BB * 128 * 4);

  // stage 0
  node_embed_kernel<<<NN * 128 / 256, 256, 0, stream>>>(x, node_w, node_b, h);
  hipMemsetAsync(loop_attr, 0, (size_t)NN * 128 * 4, stream);
  hipMemsetAsync(deg, 0, (size_t)NN * 4, stream);
  deg_loop_kernel<<<EE / 8, 256, 0, stream>>>(edge_attr, edge_w, edge_b, edge_index,
                                              loop_attr, deg);
  loop_div_kernel<<<NN * 128 / 256, 256, 0, stream>>>(loop_attr, deg);

  // layers
  for (int l = 0; l < 3; ++l) {
    conv_b_swz_kernel<<<128 * 512 / 256, 256, 0, stream>>>(lin_w + (size_t)l * 128 * 512, Bh);
    fold_watt_kernel<<<2, 256, 0, stream>>>(lin_edge_w + (size_t)l * 128 * 512,
                                            att_edge + (size_t)l * 512, watt);
    gemm_xh_kernel<<<NN / 16, 256, 0, stream>>>(h, Bh, xh);
    attn_node_kernel<<<NN / 8, 256, 0, stream>>>(xh, att_src + (size_t)l * 512,
                                                 att_dst + (size_t)l * 512, a_src, a_dst);
    a_edge_kernel<<<EN / 8, 256, 0, stream>>>(edge_attr, edge_w, edge_b, loop_attr, watt,
                                              a_edge);
    hipMemsetAsync(mx, 0, (size_t)NN * 4 * 4, stream);
    hipMemsetAsync(denom, 0, (size_t)NN * 4 * 4, stream);
    hipMemsetAsync(agg, 0, (size_t)NN * 512 * 4, stream);
    alpha_max_kernel<<<EN * 4 / 256, 256, 0, stream>>>(a_src, a_dst, a_edge, edge_index,
                                                       alpha, mx);
    exp_sum_kernel<<<EN * 4 / 256, 256, 0, stream>>>(alpha, mx, edge_index, denom);
    aggregate_kernel<<<EN * 4 / 8, 256, 0, stream>>>(alpha, denom, xh, edge_index, agg);
    finalize_kernel<<<NN * 128 / 256, 256, 0, stream>>>(agg, lin_b + (size_t)l * 128, outb);
    hipMemsetAsync(bnsum, 0, 128 * 4, stream);
    hipMemsetAsync(bnsq, 0, 128 * 4, stream);
    bn_stats_kernel<<<128, 128, 0, stream>>>(outb, bnsum, bnsq);
    bn_apply_kernel<<<NN * 128 / 256, 256, 0, stream>>>(outb, bnsum, bnsq,
                                                        bn_gamma + (size_t)l * 128,
                                                        bn_beta + (size_t)l * 128, h);
  }

  // Set2Set
  hipMemsetAsync(qstar, 0, BB * 256 * 4, stream);
  hipMemsetAsync(hhb, 0, BB * 128 * 4, stream);
  hipMemsetAsync(ccb, 0, BB * 128 * 4, stream);
  for (int s = 0; s < 3; ++s) {
    lstm_kernel<<<BB, 512, 0, stream>>>(lstm_wih, lstm_whh, lstm_b, qstar, hhb, ccb);
    s2s_attn_kernel<<<BB, 128, 0, stream>>>(h, hhb, qstar);
  }
  out_mlp_kernel<<<BB, 256, 0, stream>>>(qstar, out_w1, out_b1, out_w2, out_b2, out);

  (void)in_sizes; (void)n_in; (void)out_size; (void)ws_size;
}
